// AgentAttention_73796128079999
// MI455X (gfx1250) — compile-verified, matrix operand order checked by reading, not run on hardware
//
#include <hip/hip_runtime.h>
#include <hip/hip_bf16.h>
#include <math.h>

typedef _Float16 v16h __attribute__((ext_vector_type(16)));
typedef _Float16 v8h  __attribute__((ext_vector_type(8)));
typedef _Float16 v4h  __attribute__((ext_vector_type(4)));
typedef float    v8f  __attribute__((ext_vector_type(8)));
typedef float    f4   __attribute__((ext_vector_type(4)));

static constexpr int   BQ    = 32;
static constexpr int   NSEQ  = 784;
static constexpr int   CH    = 1024;
static constexpr int   AG    = 4;
static constexpr int   MROWS = BQ * NSEQ;           // 25088
static constexpr float SCALE = 0.03125f;            // 1024^-0.5

// CDNA5 async global->LDS copy path (ASYNCcnt), if the toolchain exposes it.
#if __has_builtin(__builtin_amdgcn_global_load_async_to_lds_b128) && \
    __has_builtin(__builtin_amdgcn_s_wait_asynccnt)
#define USE_ASYNC_LDS 1
#define GLOBAL_AS __attribute__((address_space(1)))
#define LDS_AS    __attribute__((address_space(3)))
typedef int v4i __attribute__((vector_size(16)));
typedef GLOBAL_AS v4i* gv4i_p;   // pointer to int4 in global AS
typedef LDS_AS    v4i* lv4i_p;   // pointer to int4 in LDS AS
#else
#define USE_ASYNC_LDS 0
#endif

// ---------------------------------------------------------------------------
// f32 -> f16 conversion (vectorized x4)
// ---------------------------------------------------------------------------
__global__ void cvt_f32_f16_v4(const float* __restrict__ src,
                               _Float16* __restrict__ dst, long n4) {
  long i = (long)blockIdx.x * blockDim.x + threadIdx.x;
  if (i >= n4) return;
  f4 v = ((const f4*)src)[i];
  v4h h;
  h[0] = (_Float16)v[0]; h[1] = (_Float16)v[1];
  h[2] = (_Float16)v[2]; h[3] = (_Float16)v[3];
  ((v4h*)dst)[i] = h;
}

// ---------------------------------------------------------------------------
// Tiled transpose + convert: Wt[n*K + k] = (f16) W[k*N + n]
// grid (N/32, K/32), block (32, 8)
// ---------------------------------------------------------------------------
__global__ void transpose_cvt(const float* __restrict__ W,
                              _Float16* __restrict__ Wt, int K, int N) {
  __shared__ float tile[32][33];
  int n0 = blockIdx.x * 32, k0 = blockIdx.y * 32;
  for (int j = threadIdx.y; j < 32; j += 8)
    tile[j][threadIdx.x] = W[(size_t)(k0 + j) * N + n0 + threadIdx.x];
  __syncthreads();
  for (int j = threadIdx.y; j < 32; j += 8)
    Wt[(size_t)(n0 + j) * K + k0 + threadIdx.x] = (_Float16)tile[threadIdx.x][j];
}

// ---------------------------------------------------------------------------
// WMMA GEMM:  C(MxN) = A(MxK, f16 row-major, lda) * B (given as B^T: NxK f16,
// row-major, ldb) [+ bias][gelu].  Block tile 128x128, 8 waves of wave32,
// each wave 4x2 tiles of 16x16, K stepped by 32, double-buffered LDS.
// Requires M%128==0, N%128==0, K%32==0.
// ---------------------------------------------------------------------------
#define LDSW 40   // padded row stride in halves -> conflict-free 16B frag loads

__global__ __launch_bounds__(256)
void gemm_wmma_f16(const _Float16* __restrict__ A, int lda,
                   const _Float16* __restrict__ Bt, int ldb,
                   float* __restrict__ outF, _Float16* __restrict__ outH,
                   int ldc, const float* __restrict__ bias,
                   int M, int N, int K, int act) {
  __shared__ __align__(16) _Float16 sA[2][128 * LDSW];
  __shared__ __align__(16) _Float16 sB[2][128 * LDSW];

  const int tid  = threadIdx.x;
  const int lane = tid & 31;
  const int wid  = tid >> 5;
  const int wm   = wid >> 2;       // 0..1  (M direction, 64 rows each)
  const int wn   = wid & 3;        // 0..3  (N direction, 32 cols each)
  const int lm   = lane & 15;
  const int hi   = lane >> 4;
  const int m0   = blockIdx.y * 128;
  const int n0   = blockIdx.x * 128;

  v8f acc[4][2];
#pragma unroll
  for (int i = 0; i < 4; ++i)
#pragma unroll
    for (int j = 0; j < 2; ++j)
      acc[i][j] = (v8f){0.f, 0.f, 0.f, 0.f, 0.f, 0.f, 0.f, 0.f};

  auto compute = [&](int buf) {
    v16h af[4], bf[2];
#pragma unroll
    for (int mt = 0; mt < 4; ++mt) {
      int row = wm * 64 + mt * 16 + lm;
      v8h lo = *(const v8h*)(&sA[buf][row * LDSW + hi * 8]);
      v8h h8 = *(const v8h*)(&sA[buf][row * LDSW + 16 + hi * 8]);
#pragma unroll
      for (int i = 0; i < 8; ++i) { af[mt][i] = lo[i]; af[mt][i + 8] = h8[i]; }
    }
#pragma unroll
    for (int nt = 0; nt < 2; ++nt) {
      int row = wn * 32 + nt * 16 + lm;
      v8h lo = *(const v8h*)(&sB[buf][row * LDSW + hi * 8]);
      v8h h8 = *(const v8h*)(&sB[buf][row * LDSW + 16 + hi * 8]);
#pragma unroll
      for (int i = 0; i < 8; ++i) { bf[nt][i] = lo[i]; bf[nt][i + 8] = h8[i]; }
    }
#pragma unroll
    for (int mt = 0; mt < 4; ++mt)
#pragma unroll
      for (int nt = 0; nt < 2; ++nt)
        acc[mt][nt] = __builtin_amdgcn_wmma_f32_16x16x32_f16(
            false, af[mt], false, bf[nt], (short)0, acc[mt][nt], false, false);
  };

  const int nk = K >> 5;

#if USE_ASYNC_LDS
  // -------- async global->LDS pipeline (no VGPR round-trip, ASYNCcnt) ------
  auto load_tile_async = [&](int kk, int buf) {
    const int k0 = kk * 32;
#pragma unroll
    for (int it = 0; it < 2; ++it) {
      int g   = tid + it * 256;      // 0..511
      int row = g >> 2;              // 0..127
      int ks  = (g & 3) * 8;         // 0,8,16,24
      __builtin_amdgcn_global_load_async_to_lds_b128(
          (gv4i_p)(GLOBAL_AS void*)(A + (size_t)(m0 + row) * lda + k0 + ks),
          (lv4i_p)(LDS_AS void*)(&sA[buf][row * LDSW + ks]), 0, 0);
      __builtin_amdgcn_global_load_async_to_lds_b128(
          (gv4i_p)(GLOBAL_AS void*)(Bt + (size_t)(n0 + row) * ldb + k0 + ks),
          (lv4i_p)(LDS_AS void*)(&sB[buf][row * LDSW + ks]), 0, 0);
    }
  };
  load_tile_async(0, 0);
  for (int kk = 0; kk < nk; ++kk) {
    __builtin_amdgcn_s_wait_asynccnt(0);
    __syncthreads();
    if (kk + 1 < nk) load_tile_async(kk + 1, (kk + 1) & 1);
    compute(kk & 1);
  }
#else
  // -------- register-staged software pipeline: fetch -> compute -> commit --
  auto fetch = [&](int kk, v8h* ra, v8h* rb) {
    const int k0 = kk * 32;
#pragma unroll
    for (int it = 0; it < 2; ++it) {
      int g   = tid + it * 256;
      int row = g >> 2;
      int ks  = (g & 3) * 8;
      ra[it] = *(const v8h*)(A  + (size_t)(m0 + row) * lda + k0 + ks);
      rb[it] = *(const v8h*)(Bt + (size_t)(n0 + row) * ldb + k0 + ks);
    }
  };
  auto commit = [&](int buf, const v8h* ra, const v8h* rb) {
#pragma unroll
    for (int it = 0; it < 2; ++it) {
      int g   = tid + it * 256;
      int row = g >> 2;
      int ks  = (g & 3) * 8;
      *(v8h*)(&sA[buf][row * LDSW + ks]) = ra[it];
      *(v8h*)(&sB[buf][row * LDSW + ks]) = rb[it];
    }
  };
  {
    v8h ra[2], rb[2];
    fetch(0, ra, rb);
    commit(0, ra, rb);
  }
  for (int kk = 0; kk < nk; ++kk) {
    __syncthreads();
    v8h ra[2], rb[2];
    if (kk + 1 < nk) fetch(kk + 1, ra, rb);   // global loads in flight ...
    compute(kk & 1);                          // ... hidden under 8 WMMAs
    if (kk + 1 < nk) commit((kk + 1) & 1, ra, rb);
  }
#endif

  // epilogue: C layout - VGPR r: lanes 0-15 -> M=r, lanes 16-31 -> M=8+r
#pragma unroll
  for (int mt = 0; mt < 4; ++mt) {
#pragma unroll
    for (int nt = 0; nt < 2; ++nt) {
      int col = n0 + wn * 32 + nt * 16 + lm;
      float bv = bias ? bias[col] : 0.f;
#pragma unroll
      for (int r = 0; r < 8; ++r) {
        int row = m0 + wm * 64 + mt * 16 + hi * 8 + r;
        float v = acc[mt][nt][r] + bv;
        if (act) v = 0.5f * v * (1.0f + erff(v * 0.70710678118f));
        if (outF) outF[(size_t)row * ldc + col] = v;
        if (outH) outH[(size_t)row * ldc + col] = (_Float16)v;
      }
    }
  }
}

// ---------------------------------------------------------------------------
// hbar[b, c] = mean_n h[b, n, c]   (h: MROWS x 512 f16)
// ---------------------------------------------------------------------------
__global__ void hbar_kernel(const _Float16* __restrict__ h,
                            float* __restrict__ hbar) {
  int b = blockIdx.x;
  for (int c = threadIdx.x; c < 512; c += 256) {
    float s = 0.f;
    for (int n = 0; n < NSEQ; ++n)
      s += (float)h[(size_t)(b * NSEQ + n) * 512 + c];
    hbar[b * 512 + c] = s * (1.0f / (float)NSEQ);
  }
}

// ---------------------------------------------------------------------------
// at[b, j] = hbar[b,:] @ W2[:, j] + b2[j]   (32 x 4096, K=512) -- tiny GEMM
// ---------------------------------------------------------------------------
__global__ void at_kernel(const float* __restrict__ hbar,
                          const float* __restrict__ W2,
                          const float* __restrict__ b2,
                          float* __restrict__ at) {
  __shared__ float sh[512];
  int o = blockIdx.x * 256 + threadIdx.x;   // < 32*4096
  int b = o >> 12, j = o & 4095;
  for (int k = threadIdx.x; k < 512; k += 256) sh[k] = hbar[b * 512 + k];
  __syncthreads();
  float s = b2[j];
  for (int k = 0; k < 512; ++k) s += sh[k] * W2[(size_t)k * 4096 + j];
  at[o] = s;
}

// ---------------------------------------------------------------------------
// Stage 1: per (b,a): softmax_n(SCALE * at . k[n]), write agent_rep,
//          agent_v[d] = sum_n p[n] * v[n,d].   One block per (b,a).
// ---------------------------------------------------------------------------
__global__ __launch_bounds__(256)
void stage1_kernel(const _Float16* __restrict__ qkv, const float* __restrict__ at,
                   float* __restrict__ agent_v, float* __restrict__ rep) {
  __shared__ float s_at[CH];
  __shared__ float s_p[NSEQ];
  __shared__ float s_red[256];
  const int ba = blockIdx.x, b = ba >> 2;
  const int tid = threadIdx.x, lane = tid & 31, wid = tid >> 5;

  for (int d = tid; d < CH; d += 256) s_at[d] = at[(size_t)ba * CH + d];
  __syncthreads();

  for (int n = wid; n < NSEQ; n += 8) {
    const _Float16* kp = qkv + (size_t)(b * NSEQ + n) * 3072 + CH + lane * 32;
    float s = 0.f;
#pragma unroll
    for (int j8 = 0; j8 < 4; ++j8) {
      v8h kv = *(const v8h*)(kp + j8 * 8);
#pragma unroll
      for (int j = 0; j < 8; ++j)
        s += s_at[lane * 32 + j8 * 8 + j] * (float)kv[j];
    }
#pragma unroll
    for (int off = 16; off > 0; off >>= 1) s += __shfl_xor(s, off, 32);
    if (lane == 0) s_p[n] = s * SCALE;
  }
  __syncthreads();

  float lm = -1e30f;
  for (int n = tid; n < NSEQ; n += 256) lm = fmaxf(lm, s_p[n]);
  s_red[tid] = lm; __syncthreads();
  for (int st = 128; st > 0; st >>= 1) {
    if (tid < st) s_red[tid] = fmaxf(s_red[tid], s_red[tid + st]);
    __syncthreads();
  }
  float mx = s_red[0]; __syncthreads();

  float ls = 0.f;
  for (int n = tid; n < NSEQ; n += 256) {
    float e = expf(s_p[n] - mx); s_p[n] = e; ls += e;
  }
  s_red[tid] = ls; __syncthreads();
  for (int st = 128; st > 0; st >>= 1) {
    if (tid < st) s_red[tid] += s_red[tid + st];
    __syncthreads();
  }
  float inv = 1.0f / s_red[0]; __syncthreads();

  for (int n = tid; n < NSEQ; n += 256) {
    float p = s_p[n] * inv; s_p[n] = p;
    rep[(size_t)ba * NSEQ + n] = p;
  }
  __syncthreads();

  // agent_v: each thread owns 4 consecutive d (8B v4h loads, fully coalesced)
  {
    const int d4 = tid * 4;                      // 256 threads x 4 = 1024
    float a0 = 0.f, a1 = 0.f, a2 = 0.f, a3 = 0.f;
    for (int n = 0; n < NSEQ; ++n) {
      float p = s_p[n];
      v4h vv = *(const v4h*)(qkv + (size_t)(b * NSEQ + n) * 3072 + 2048 + d4);
      a0 += p * (float)vv[0]; a1 += p * (float)vv[1];
      a2 += p * (float)vv[2]; a3 += p * (float)vv[3];
    }
    f4 o; o[0] = a0; o[1] = a1; o[2] = a2; o[3] = a3;
    *(f4*)(agent_v + (size_t)ba * CH + d4) = o;
  }
}

// ---------------------------------------------------------------------------
// Stage 2: per row (b,n): softmax_a(SCALE * q . at[a]), out = sum_a p_a*agent_v[a].
// One wave per row, 8 waves per block.
// ---------------------------------------------------------------------------
__global__ __launch_bounds__(256)
void stage2_kernel(const _Float16* __restrict__ qkv, const float* __restrict__ at,
                   const float* __restrict__ agent_v,
                   _Float16* __restrict__ attn_out) {
  const int tid = threadIdx.x, lane = tid & 31, wid = tid >> 5;
  const int r = blockIdx.x * 8 + wid;      // grid sized exactly: r < MROWS
  const int b = r / NSEQ;

  const _Float16* qp = qkv + (size_t)r * 3072 + lane * 32;
  float qv[32];
#pragma unroll
  for (int j8 = 0; j8 < 4; ++j8) {
    v8h q8 = *(const v8h*)(qp + j8 * 8);
#pragma unroll
    for (int j = 0; j < 8; ++j) qv[j8 * 8 + j] = (float)q8[j];
  }

  float lg[4];
#pragma unroll
  for (int a = 0; a < 4; ++a) {
    const float* ap = at + (size_t)(b * 4 + a) * CH + lane * 32;
    float s = 0.f;
#pragma unroll
    for (int j = 0; j < 32; ++j) s += qv[j] * ap[j];
#pragma unroll
    for (int off = 16; off > 0; off >>= 1) s += __shfl_xor(s, off, 32);
    lg[a] = s * SCALE;
  }
  float mx = fmaxf(fmaxf(lg[0], lg[1]), fmaxf(lg[2], lg[3]));
  float p[4], sum = 0.f;
#pragma unroll
  for (int a = 0; a < 4; ++a) { p[a] = expf(lg[a] - mx); sum += p[a]; }
  float inv = 1.0f / sum;

  float o[32];
#pragma unroll
  for (int j = 0; j < 32; ++j) o[j] = 0.f;
#pragma unroll
  for (int a = 0; a < 4; ++a) {
    float pa = p[a] * inv;
    const float* vp = agent_v + (size_t)(b * 4 + a) * CH + lane * 32;
#pragma unroll
    for (int j = 0; j < 32; ++j) o[j] += pa * vp[j];
  }
  _Float16* op = attn_out + (size_t)r * CH + lane * 32;
#pragma unroll
  for (int j8 = 0; j8 < 4; ++j8) {
    v8h w;
#pragma unroll
    for (int j = 0; j < 8; ++j) w[j] = (_Float16)o[j8 * 8 + j];
    *(v8h*)(op + j8 * 8) = w;
  }
}

// ---------------------------------------------------------------------------
extern "C" void kernel_launch(void* const* d_in, const int* in_sizes, int n_in,
                              void* d_out, int out_size, void* d_ws, size_t ws_size,
                              hipStream_t stream) {
  const float* x     = (const float*)d_in[0];
  const float* W_qkv = (const float*)d_in[1];
  const float* W1    = (const float*)d_in[2];
  const float* b1    = (const float*)d_in[3];
  const float* W2    = (const float*)d_in[4];
  const float* b2    = (const float*)d_in[5];
  const float* Wp    = (const float*)d_in[6];
  const float* bp    = (const float*)d_in[7];
  float* out = (float*)d_out;
  float* rep = out + (size_t)MROWS * CH;   // agent_rep tail: (32,1,4,784)

  // workspace carve-up (all 256B aligned)
  char* ws = (char*)d_ws;
  size_t off = 0;
  auto carve = [&](size_t bytes) { void* p = ws + off; off += (bytes + 255) & ~(size_t)255; return p; };
  _Float16* x_h    = (_Float16*)carve((size_t)MROWS * CH * 2);        // 51.4MB
  _Float16* WqkvT  = (_Float16*)carve((size_t)3072 * 1024 * 2);       // 6.3MB
  _Float16* W1T    = (_Float16*)carve((size_t)512 * 1024 * 2);        // 1.0MB
  _Float16* WpT    = (_Float16*)carve((size_t)1024 * 1024 * 2);       // 2.1MB
  _Float16* qkv_h  = (_Float16*)carve((size_t)MROWS * 3072 * 2);      // 154MB
  _Float16* h_h    = (_Float16*)carve((size_t)MROWS * 512 * 2);       // 25.7MB
  float*    hbar   = (float*)carve((size_t)32 * 512 * 4);
  float*    at     = (float*)carve((size_t)32 * 4096 * 4);
  float*    agv    = (float*)carve((size_t)32 * 4096 * 4);
  _Float16* attn_h = (_Float16*)carve((size_t)MROWS * CH * 2);        // 51.4MB

  // 1) convert x -> f16
  {
    long n4 = (long)MROWS * CH / 4;
    cvt_f32_f16_v4<<<(n4 + 255) / 256, 256, 0, stream>>>(x, x_h, n4);
  }
  // 2) transpose+convert weights (B^T layout for WMMA B operand)
  transpose_cvt<<<dim3(3072 / 32, 1024 / 32), dim3(32, 8), 0, stream>>>(W_qkv, WqkvT, 1024, 3072);
  transpose_cvt<<<dim3(512 / 32, 1024 / 32), dim3(32, 8), 0, stream>>>(W1, W1T, 1024, 512);
  transpose_cvt<<<dim3(1024 / 32, 1024 / 32), dim3(32, 8), 0, stream>>>(Wp, WpT, 1024, 1024);

  // 3) qkv = x @ W_qkv  (f16 out, no bias)
  gemm_wmma_f16<<<dim3(3072 / 128, MROWS / 128), 256, 0, stream>>>(
      x_h, CH, WqkvT, CH, nullptr, qkv_h, 3072, nullptr, MROWS, 3072, CH, 0);

  // 4) h = gelu(q @ W1 + b1)  (q = first 1024 cols of qkv, lda=3072)
  gemm_wmma_f16<<<dim3(512 / 128, MROWS / 128), 256, 0, stream>>>(
      qkv_h, 3072, W1T, CH, nullptr, h_h, 512, b1, MROWS, 512, CH, 1);

  // 5) hbar = mean_n h ; at = hbar @ W2 + b2  (mean commutes with linear map)
  hbar_kernel<<<32, 256, 0, stream>>>(h_h, hbar);
  at_kernel<<<(32 * 4096) / 256, 256, 0, stream>>>(hbar, W2, b2, at);

  // 6) stage 1: agent attention over keys -> rep + agent_v
  stage1_kernel<<<32 * AG, 256, 0, stream>>>(qkv_h, at, agv, rep);

  // 7) stage 2: queries attend to agents -> attn_h (f16)
  stage2_kernel<<<MROWS / 8, 256, 0, stream>>>(qkv_h, at, agv, attn_h);

  // 8) out = attn @ Wp + bp  (f32 out to d_out)
  gemm_wmma_f16<<<dim3(1024 / 128, MROWS / 128), 256, 0, stream>>>(
      attn_h, CH, WpT, CH, out, nullptr, CH, bp, MROWS, CH, CH, 0);

  (void)in_sizes; (void)n_in; (void)out_size; (void)ws_size;
}
